// RelativePoseHead_19980187861737
// MI455X (gfx1250) — matrix operand order repaired and checked by reading, not compiled, not run on hardware
//
#include <hip/hip_runtime.h>
#include <math.h>

#define NB  64
#define SQ  2048
#define HH  1024
#define DDIM 256
#define NHH 8

typedef __attribute__((ext_vector_type(16))) _Float16 v16h;
typedef __attribute__((ext_vector_type(8)))  float    v8f;
typedef __attribute__((ext_vector_type(4)))  unsigned int u32x4;
typedef __attribute__((ext_vector_type(8)))  int      i32x8;
typedef __attribute__((ext_vector_type(4)))  int      i32x4;

// ---------------------------------------------------------------------------
// Tensor Data Mover: 2D tile (rows x cols of f32) global -> LDS.
// D# built per CDNA5 ISA ch.8: group0 = {count=1, lds_addr, global_addr, type=2},
// group1 = {data_size=4B, tensor_dim0/1, tile_dim0/1, tensor_dim0_stride}.
// Issue from ONE wave only (TDM ignores EXEC, issues per-wave).
// ---------------------------------------------------------------------------
__device__ __forceinline__ unsigned lds_addr_of(const void* p) {
  // generic pointers to LDS carry the wave-relative LDS offset in the low 32 bits
  return (unsigned)(unsigned long long)p;
}

__device__ __forceinline__ void tdm_load_2d_f32(unsigned lds_addr, const void* gsrc,
                                                unsigned tensor_d0, unsigned tensor_d1,
                                                unsigned tile_d0, unsigned tile_d1,
                                                unsigned stride_d0) {
  const unsigned long long ga = (unsigned long long)gsrc;
  u32x4 g0;
  g0[0] = 1u;                                            // count=1, user descriptor
  g0[1] = lds_addr;                                      // LDS byte address
  g0[2] = (unsigned)(ga & 0xFFFFFFFFu);                  // global_addr[31:0]
  g0[3] = (unsigned)((ga >> 32) & 0x01FFFFFFu)           // global_addr[56:32]
        | (2u << 30);                                    // type=2 ("image")
  i32x8 g1;
  g1[0] = (int)(2u << 16);                               // data_size=2 (4 bytes)
  g1[1] = (int)((tensor_d0 & 0xFFFFu) << 16);            // tensor_dim0[15:0]
  g1[2] = (int)((tensor_d0 >> 16) | ((tensor_d1 & 0xFFFFu) << 16)); // d0 hi | d1 lo
  g1[3] = (int)((tensor_d1 >> 16) | (tile_d0 << 16));    // d1 hi | tile_dim0
  g1[4] = (int)(tile_d1 & 0xFFFFu);                      // tile_dim1 | tile_dim2=0
  g1[5] = (int)stride_d0;                                // tensor_dim0_stride[31:0]
  g1[6] = 0;                                             // stride hi | dim1_stride lo
  g1[7] = 0;
  const i32x4 z4 = {0, 0, 0, 0};
#if defined(__clang_major__) && (__clang_major__ >= 23)
  const i32x8 z8 = {0, 0, 0, 0, 0, 0, 0, 0};
  __builtin_amdgcn_tensor_load_to_lds(g0, g1, z4, z4, z8, 0);
#else
  __builtin_amdgcn_tensor_load_to_lds(g0, g1, z4, z4, 0);
#endif
}

// ---------------------------------------------------------------------------
// WMMA helpers (wave32, 16x16 tiles, K-step 32, f16 operands / f32 accum)
// A layout (ISA 7.12.2): lane l: M=l&15; e<8 -> K=k0+hi*8+e, e>=8 -> K=k0+16+hi*8+(e-8).
// B layout (32x16): lane l: N=l&15; e<8 -> K=k0+hi*8+e, e>=8 -> K=k0+16+hi*8+(e-8).
// C/D v8f: lane l: N=l&15, VGPR r -> M=r+8*hi.
// ---------------------------------------------------------------------------
__device__ __forceinline__ v8f wmma_tile_f16(const float* __restrict__ Xb, int ldx,
                                             const float* __restrict__ Wb, int ldw,
                                             int K, int lane) {
  const int mr = lane & 15;
  const int hi = lane >> 4;
  const float* xa = Xb + (size_t)mr * ldx;
  const float* wb = Wb + (size_t)mr * ldw;
  v8f acc = {};
#pragma unroll 2
  for (int k0 = 0; k0 < K; k0 += 32) {
    v16h A, Bt;
    const float* pa = xa + k0 + hi * 8;
#pragma unroll
    for (int e = 0; e < 8; ++e) { A[e] = (_Float16)pa[e]; A[e + 8] = (_Float16)pa[16 + e]; }
    const float* pb = wb + k0 + hi * 8;
#pragma unroll
    for (int e = 0; e < 8; ++e) { Bt[e] = (_Float16)pb[e]; Bt[e + 8] = (_Float16)pb[16 + e]; }
    acc = __builtin_amdgcn_wmma_f32_16x16x32_f16(false, A, false, Bt, (short)0, acc,
                                                 false, false);
  }
  return acc;
}

__device__ __forceinline__ void wmma_store(v8f acc, float* __restrict__ Yb, int ldy,
                                           const float* __restrict__ bias, int lane, int epi) {
  const int nn = lane & 15;
  const int hi = lane >> 4;
  const float bvv = bias ? bias[nn] : 0.f;
#pragma unroll
  for (int r = 0; r < 8; ++r) {
    float vv = acc[r] + bvv;
    if (epi == 1) vv = 0.5f * vv * (1.f + erff(vv * 0.7071067811865476f)); // exact GELU
    Yb[(size_t)(r + 8 * hi) * ldy + nn] = vv;
  }
}

// Generic Y[M,N] = X[M,K] @ W[N,K]^T + bias, optional GELU. One wave = one 16x16 tile.
__global__ void wmma_gemm_kernel(const float* __restrict__ X, int ldx,
                                 const float* __restrict__ W, int ldw,
                                 const float* __restrict__ bias,
                                 float* __restrict__ Y, int ldy,
                                 int Mtiles, int Ntiles, int K, int epi) {
  const int wave = threadIdx.x >> 5, lane = threadIdx.x & 31;
  const int t = blockIdx.x * (blockDim.x >> 5) + wave;
  if (t >= Mtiles * Ntiles) return;
  const int m0 = (t / Ntiles) * 16, n0 = (t % Ntiles) * 16;
  v8f acc = wmma_tile_f16(X + (size_t)m0 * ldx, ldx, W + (size_t)n0 * ldw, ldw, K, lane);
  wmma_store(acc, Y + (size_t)m0 * ldy + n0, ldy, bias ? bias + n0 : nullptr, lane, epi);
}

// Head-sliced GEMM: O[b, h*32+n] = Wv'[h*32+n,:] . a[b,h,:] + bo. 8 blocks (1/head), 8 waves.
__global__ void wmma_gemm_o_kernel(const float* __restrict__ a,     // [64][8][1024]
                                   const float* __restrict__ Wvp,   // [256][1024]
                                   const float* __restrict__ bo,    // [256]
                                   float* __restrict__ O) {         // [64][256]
  const int h = blockIdx.x;
  const int wave = threadIdx.x >> 5, lane = threadIdx.x & 31;
  const int m0 = (wave >> 1) * 16;            // 4 m-tiles (batch rows)
  const int n0 = h * 32 + (wave & 1) * 16;    // 2 n-tiles per head
  const float* X = a + (size_t)h * HH + (size_t)m0 * (NHH * HH);   // row stride 8192
  const float* W = Wvp + (size_t)n0 * HH;
  v8f acc = wmma_tile_f16(X, NHH * HH, W, HH, HH, lane);
  wmma_store(acc, O + (size_t)m0 * DDIM + n0, DDIM, bo + n0, lane, 0);
}

// ---------------------------------------------------------------------------
// Prep kernels (tiny)
// ---------------------------------------------------------------------------
__global__ void prep1_kernel(const float* __restrict__ lq, const float* __restrict__ in_w,
                             const float* __restrict__ in_b, const float* __restrict__ proj_w,
                             const float* __restrict__ proj_b,
                             float* __restrict__ q, float* __restrict__ svD,
                             float* __restrict__ c, float* __restrict__ sv,
                             float* __restrict__ bo) {
  __shared__ float qs[DDIM];
  __shared__ float sds[NHH * DDIM];
  const int tid = threadIdx.x;
  {
    const float* wq = in_w;
    float s = in_b[tid];
    for (int j = 0; j < DDIM; ++j) s += lq[j] * wq[tid * DDIM + j];
    qs[tid] = s; q[tid] = s;
  }
  {
    const float* wv = in_w + 512 * DDIM;
    float s = in_b[512 + tid];
    for (int j = 0; j < DDIM; ++j) s += wv[tid * DDIM + j] * proj_b[j];
    bo[tid] = s;
  }
  __syncthreads();
  const float* wk = in_w + 256 * DDIM;
  for (int e = tid; e < NHH * DDIM; e += 256) {
    const int h = e >> 8, i = e & 255;
    float acc = 0.f;
    for (int dd = 0; dd < 32; ++dd) acc += qs[h * 32 + dd] * wk[(h * 32 + dd) * DDIM + i];
    sds[e] = acc; svD[e] = acc;
  }
  __syncthreads();
  if (tid < NHH) {
    const float* bk = in_b + 256;
    float acc = 0.f;
    for (int dd = 0; dd < 32; ++dd) acc += qs[tid * 32 + dd] * bk[tid * 32 + dd];
    for (int i = 0; i < DDIM; ++i) acc += sds[tid * DDIM + i] * proj_b[i];
    c[tid] = acc;
  }
  for (int e = tid; e < NHH * HH; e += 256) {
    const int h = e >> 10, j = e & 1023;
    float acc = 0.f;
    for (int i = 0; i < DDIM; ++i) acc += sds[h * DDIM + i] * proj_w[i * HH + j];
    sv[e] = acc;
  }
}

__global__ void prep_wvp_kernel(const float* __restrict__ in_w,
                                const float* __restrict__ proj_w,
                                float* __restrict__ Wvp) {
  const int idx = blockIdx.x * 256 + threadIdx.x;
  const int r = idx >> 10, j = idx & 1023;
  const float* wv = in_w + 512 * DDIM + r * DDIM;
  float acc = 0.f;
  for (int i = 0; i < DDIM; ++i) acc += wv[i] * proj_w[i * HH + j];
  Wvp[idx] = acc;
}

// ---------------------------------------------------------------------------
// Heavy pass 1: scores[b,h,s] = scale * (feats[b,s,:] . sv[h,:] + c[h])
// sv table staged by TDM (one tensor_load_to_lds per block, wave 0 only).
// ---------------------------------------------------------------------------
__global__ void scores_kernel(const float* __restrict__ feats,
                              const float* __restrict__ sv,
                              const float* __restrict__ c,
                              float* __restrict__ scores) {
  __shared__ float lsv[NHH * HH];   // 32 KB
  __shared__ float lc[NHH];
  const int b = blockIdx.y;
  if (threadIdx.x < 32) {           // wave 0 issues the DMA; TDM ignores EXEC
    tdm_load_2d_f32(lds_addr_of(lsv), sv, HH, NHH, HH, NHH, HH);
    if (threadIdx.x < NHH) lc[threadIdx.x] = c[threadIdx.x];
    __builtin_amdgcn_s_wait_tensorcnt(0);
  }
  __syncthreads();
  const int wave = threadIdx.x >> 5, lane = threadIdx.x & 31;
  const int s0 = blockIdx.x * 64 + wave * 8;
  const float* fb = feats + (size_t)b * SQ * HH;
  for (int r = 0; r < 8; ++r) {
    const int s = s0 + r;
    const float* rowf = fb + (size_t)s * HH;
    if (s + 1 < SQ) __builtin_prefetch(rowf + HH, 0, 1);   // global_prefetch_b8
    const float4* row4 = (const float4*)rowf;
    float acc[NHH];
#pragma unroll
    for (int h = 0; h < NHH; ++h) acc[h] = 0.f;
    for (int it = 0; it < 8; ++it) {
      const float4 x = row4[it * 32 + lane];
#pragma unroll
      for (int h = 0; h < NHH; ++h) {
        const float4 w = *(const float4*)&lsv[h * HH + it * 128 + lane * 4];
        acc[h] += x.x * w.x + x.y * w.y + x.z * w.z + x.w * w.w;
      }
    }
#pragma unroll
    for (int h = 0; h < NHH; ++h)
      for (int off = 16; off > 0; off >>= 1)
        acc[h] += __shfl_xor(acc[h], off, 32);   // wave32 reduction
    if (lane == 0) {
#pragma unroll
      for (int h = 0; h < NHH; ++h)
        scores[((size_t)b * NHH + h) * SQ + s] = 0.17677669529663687f * (acc[h] + lc[h]);
    }
  }
}

// Softmax over s per (b,h). 512 blocks.
__global__ void softmax_kernel(float* __restrict__ scores) {
  float* p = scores + (size_t)blockIdx.x * SQ;
  __shared__ float red[256];
  const int tid = threadIdx.x;
  float m = -3.4e38f;
  for (int i = tid; i < SQ; i += 256) m = fmaxf(m, p[i]);
  red[tid] = m; __syncthreads();
  for (int st = 128; st > 0; st >>= 1) { if (tid < st) red[tid] = fmaxf(red[tid], red[tid + st]); __syncthreads(); }
  m = red[0]; __syncthreads();
  float sum = 0.f;
  for (int i = tid; i < SQ; i += 256) { const float e = expf(p[i] - m); p[i] = e; sum += e; }
  red[tid] = sum; __syncthreads();
  for (int st = 128; st > 0; st >>= 1) { if (tid < st) red[tid] += red[tid + st]; __syncthreads(); }
  const float inv = 1.f / red[0];
  for (int i = tid; i < SQ; i += 256) p[i] *= inv;
}

// ---------------------------------------------------------------------------
// Heavy pass 2: partial a[b,h,:] = sum_s attn[b,h,s]*feats[b,s,:]
// attn tile (8 rows x 256, row-stride 2048) staged by a single 2D TDM load.
// Deterministic (no float atomics): partials + reduce.
// ---------------------------------------------------------------------------
__global__ void wsum_kernel(const float* __restrict__ feats,
                            const float* __restrict__ attn,
                            float* __restrict__ ap) {   // [64][8][8][1024]
  __shared__ float lat[NHH][256];
  const int b = blockIdx.y, sc = blockIdx.x;
  const int tid = threadIdx.x;
  if (tid < 32) {                   // wave 0 issues the DMA
    const float* src = attn + (size_t)b * NHH * SQ + sc * 256;
    tdm_load_2d_f32(lds_addr_of(lat), src, SQ, NHH, 256, NHH, SQ);
    __builtin_amdgcn_s_wait_tensorcnt(0);
  }
  __syncthreads();
  float acc[NHH][4];
#pragma unroll
  for (int h = 0; h < NHH; ++h) { acc[h][0] = acc[h][1] = acc[h][2] = acc[h][3] = 0.f; }
  const float4* fb = (const float4*)(feats + ((size_t)b * SQ + sc * 256) * HH);
  for (int sp = 0; sp < 256; ++sp) {
    const float4 x = fb[(size_t)sp * 256 + tid];
#pragma unroll
    for (int h = 0; h < NHH; ++h) {
      const float w = lat[h][sp];
      acc[h][0] += w * x.x; acc[h][1] += w * x.y; acc[h][2] += w * x.z; acc[h][3] += w * x.w;
    }
  }
  float* apb = ap + ((size_t)b * 8 + sc) * (NHH * HH) + tid * 4;
#pragma unroll
  for (int h = 0; h < NHH; ++h)
    *(float4*)(apb + (size_t)h * HH) = make_float4(acc[h][0], acc[h][1], acc[h][2], acc[h][3]);
}

__global__ void reduce_a_kernel(const float* __restrict__ ap, float* __restrict__ a) {
  const int idx = blockIdx.x * 256 + threadIdx.x;   // 524288
  const int b = idx >> 13;
  const int hj = idx & 8191;
  float s = 0.f;
#pragma unroll
  for (int sc = 0; sc < 8; ++sc) s += ap[((size_t)b * 8 + sc) * (NHH * HH) + hj];
  a[idx] = s;
}

// ---------------------------------------------------------------------------
// LayerNorms + head
// ---------------------------------------------------------------------------
__global__ void ln1_kernel(const float* __restrict__ ao, const float* __restrict__ lq,
                           const float* __restrict__ w, const float* __restrict__ bb,
                           float* __restrict__ x) {
  const int b = blockIdx.x, d = threadIdx.x;
  __shared__ float red[256];
  const float v = lq[d] + ao[b * DDIM + d];
  red[d] = v; __syncthreads();
  for (int st = 128; st > 0; st >>= 1) { if (d < st) red[d] += red[d + st]; __syncthreads(); }
  const float mu = red[0] * (1.f / 256.f); __syncthreads();
  const float dv = v - mu;
  red[d] = dv * dv; __syncthreads();
  for (int st = 128; st > 0; st >>= 1) { if (d < st) red[d] += red[d + st]; __syncthreads(); }
  const float var = red[0] * (1.f / 256.f);
  x[b * DDIM + d] = dv * rsqrtf(var + 1e-5f) * w[d] + bb[d];
}

__global__ void ln2_head_kernel(const float* __restrict__ x, const float* __restrict__ y,
                                const float* __restrict__ w, const float* __restrict__ bb,
                                const float* __restrict__ hw, const float* __restrict__ hb,
                                float* __restrict__ pose) {
  const int b = blockIdx.x, d = threadIdx.x;
  __shared__ float red[256];
  __shared__ float xs[256];
  const float v = x[b * DDIM + d] + y[b * DDIM + d];
  red[d] = v; __syncthreads();
  for (int st = 128; st > 0; st >>= 1) { if (d < st) red[d] += red[d + st]; __syncthreads(); }
  const float mu = red[0] * (1.f / 256.f); __syncthreads();
  const float dv = v - mu;
  red[d] = dv * dv; __syncthreads();
  for (int st = 128; st > 0; st >>= 1) { if (d < st) red[d] += red[d + st]; __syncthreads(); }
  const float var = red[0] * (1.f / 256.f);
  xs[d] = dv * rsqrtf(var + 1e-5f) * w[d] + bb[d];
  __syncthreads();
  if (d < 12) {
    float s = hb[d];
    for (int i = 0; i < DDIM; ++i) s += hw[d * DDIM + i] * xs[i];
    pose[b * 12 + d] = s;
  }
}

// ---------------------------------------------------------------------------
// 3x3 nearest-rotation (matches reference SVD orthogonalization). 1 thread/matrix.
// ---------------------------------------------------------------------------
__device__ __forceinline__ float det3(const float M[3][3]) {
  return M[0][0] * (M[1][1] * M[2][2] - M[1][2] * M[2][1])
       - M[0][1] * (M[1][0] * M[2][2] - M[1][2] * M[2][0])
       + M[0][2] * (M[1][0] * M[2][1] - M[1][1] * M[2][0]);
}

__global__ void svd_head_kernel(const float* __restrict__ pose, float* __restrict__ out) {
  const int b = threadIdx.x;
  if (b >= NB) return;
  const float* p = pose + b * 12;
  float A[3][3];
  for (int i = 0; i < 3; ++i) {
    const float r0 = p[i * 3 + 0], r1 = p[i * 3 + 1], r2 = p[i * 3 + 2];
    const float inv = 1.f / fmaxf(sqrtf(r0 * r0 + r1 * r1 + r2 * r2), 1e-12f);
    A[i][0] = r0 * inv; A[i][1] = r1 * inv; A[i][2] = r2 * inv;
  }
  float Bm[3][3];
  for (int i = 0; i < 3; ++i)
    for (int j = 0; j < 3; ++j) {
      float s = 0.f;
      for (int k = 0; k < 3; ++k) s += A[k][i] * A[k][j];
      Bm[i][j] = s;
    }
  float V[3][3] = {{1, 0, 0}, {0, 1, 0}, {0, 0, 1}};
  for (int sweep = 0; sweep < 10; ++sweep) {
    for (int pair = 0; pair < 3; ++pair) {
      const int pp = (pair == 2) ? 1 : 0;
      const int qq = (pair == 0) ? 1 : 2;
      const float apq = Bm[pp][qq];
      if (fabsf(apq) < 1e-13f) continue;
      const float tau = (Bm[qq][qq] - Bm[pp][pp]) / (2.f * apq);
      const float t = ((tau >= 0.f) ? 1.f : -1.f) / (fabsf(tau) + sqrtf(1.f + tau * tau));
      const float cc = rsqrtf(1.f + t * t);
      const float ss = t * cc;
      for (int k = 0; k < 3; ++k) { const float u = Bm[k][pp], w = Bm[k][qq]; Bm[k][pp] = cc * u - ss * w; Bm[k][qq] = ss * u + cc * w; }
      for (int k = 0; k < 3; ++k) { const float u = Bm[pp][k], w = Bm[qq][k]; Bm[pp][k] = cc * u - ss * w; Bm[qq][k] = ss * u + cc * w; }
      for (int k = 0; k < 3; ++k) { const float u = V[k][pp], w = V[k][qq]; V[k][pp] = cc * u - ss * w; V[k][qq] = ss * u + cc * w; }
    }
  }
  float lam[3] = {Bm[0][0], Bm[1][1], Bm[2][2]};
  for (int i = 0; i < 2; ++i)
    for (int j = i + 1; j < 3; ++j)
      if (lam[j] > lam[i]) {
        const float tl = lam[i]; lam[i] = lam[j]; lam[j] = tl;
        for (int k = 0; k < 3; ++k) { const float tv = V[k][i]; V[k][i] = V[k][j]; V[k][j] = tv; }
      }
  float U[3][3];
  for (int i = 0; i < 3; ++i) {
    float u0 = 0, u1 = 0, u2 = 0;
    for (int k = 0; k < 3; ++k) { u0 += A[0][k] * V[k][i]; u1 += A[1][k] * V[k][i]; u2 += A[2][k] * V[k][i]; }
    const float nrm = sqrtf(u0 * u0 + u1 * u1 + u2 * u2);
    if (i < 2 || nrm > 1e-6f) {
      const float inv = 1.f / fmaxf(nrm, 1e-12f);
      U[0][i] = u0 * inv; U[1][i] = u1 * inv; U[2][i] = u2 * inv;
    } else {   // rank-deficient fallback: u2 = u0 x u1
      U[0][2] = U[1][0] * U[2][1] - U[2][0] * U[1][1];
      U[1][2] = U[2][0] * U[0][1] - U[0][0] * U[2][1];
      U[2][2] = U[0][0] * U[1][1] - U[1][0] * U[0][1];
    }
  }
  const float dd = (det3(U) * det3(V) >= 0.f) ? 1.f : -1.f;
  float* o = out + b * 12;
  for (int i = 0; i < 3; ++i)
    for (int j = 0; j < 3; ++j)
      o[i * 3 + j] = U[i][0] * V[j][0] + U[i][1] * V[j][1] + dd * U[i][2] * V[j][2];
  o[9] = p[9]; o[10] = p[10]; o[11] = p[11];
}

// ---------------------------------------------------------------------------
// Launch
// ---------------------------------------------------------------------------
extern "C" void kernel_launch(void* const* d_in, const int* in_sizes, int n_in,
                              void* d_out, int out_size, void* d_ws, size_t ws_size,
                              hipStream_t stream) {
  const float* feats      = (const float*)d_in[0];
  const float* lq         = (const float*)d_in[1];
  const float* proj_w     = (const float*)d_in[2];
  const float* proj_b     = (const float*)d_in[3];
  const float* in_proj_w  = (const float*)d_in[4];
  const float* in_proj_b  = (const float*)d_in[5];
  const float* out_proj_w = (const float*)d_in[6];
  const float* out_proj_b = (const float*)d_in[7];
  const float* ln1_w = (const float*)d_in[8];
  const float* ln1_b = (const float*)d_in[9];
  const float* ln2_w = (const float*)d_in[10];
  const float* ln2_b = (const float*)d_in[11];
  const float* mlp_w1 = (const float*)d_in[12];
  const float* mlp_b1 = (const float*)d_in[13];
  const float* mlp_w2 = (const float*)d_in[14];
  const float* mlp_b2 = (const float*)d_in[15];
  const float* head_w = (const float*)d_in[16];
  const float* head_b = (const float*)d_in[17];

  // workspace layout (float elements)
  const size_t Q_OFF   = 0;
  const size_t SVD_OFF = 256;
  const size_t C_OFF   = 2304;
  const size_t SV_OFF  = 2560;
  const size_t BO_OFF  = 10752;
  const size_t WVP_OFF = 11264;
  const size_t SC_OFF  = 273408;
  const size_t AP_OFF  = 1321984;
  const size_t A_OFF   = 5516288;
  const size_t O_OFF   = 6040576;
  const size_t AO_OFF  = 6056960;
  const size_t X_OFF   = 6073344;
  const size_t G_OFF   = 6089728;
  const size_t Y_OFF   = 6155264;
  const size_t P_OFF   = 6171648;
  const size_t WS_TOTAL = 6172416;
  if (ws_size < WS_TOTAL * sizeof(float)) return;

  float* ws = (float*)d_ws;
  float* q   = ws + Q_OFF;   float* svD = ws + SVD_OFF; float* c  = ws + C_OFF;
  float* sv  = ws + SV_OFF;  float* bo  = ws + BO_OFF;  float* Wvp = ws + WVP_OFF;
  float* sc  = ws + SC_OFF;  float* ap  = ws + AP_OFF;  float* a  = ws + A_OFF;
  float* O   = ws + O_OFF;   float* AO  = ws + AO_OFF;  float* X  = ws + X_OFF;
  float* G   = ws + G_OFF;   float* Y   = ws + Y_OFF;   float* pose = ws + P_OFF;

  // folded weights for the rank-1-query attention
  prep1_kernel<<<1, 256, 0, stream>>>(lq, in_proj_w, in_proj_b, proj_w, proj_b,
                                      q, svD, c, sv, bo);
  prep_wvp_kernel<<<1024, 256, 0, stream>>>(in_proj_w, proj_w, Wvp);

  // pass 1 over feats: scores, then softmax
  scores_kernel<<<dim3(32, NB), 256, 0, stream>>>(feats, sv, c, sc);
  softmax_kernel<<<NB * NHH, 256, 0, stream>>>(sc);

  // pass 2 over feats: attention-weighted feature sums (deterministic two-stage)
  wsum_kernel<<<dim3(8, NB), 256, 0, stream>>>(feats, sc, ap);
  reduce_a_kernel<<<2048, 256, 0, stream>>>(ap, a);

  // transformer block on [64 x 256] via WMMA f16 (f32 accumulate)
  wmma_gemm_o_kernel<<<NHH, 256, 0, stream>>>(a, Wvp, bo, O);
  wmma_gemm_kernel<<<8, 256, 0, stream>>>(O, DDIM, out_proj_w, DDIM, out_proj_b,
                                          AO, DDIM, 4, 16, DDIM, 0);
  ln1_kernel<<<NB, 256, 0, stream>>>(AO, lq, ln1_w, ln1_b, X);
  wmma_gemm_kernel<<<32, 256, 0, stream>>>(X, DDIM, mlp_w1, DDIM, mlp_b1,
                                           G, HH, 4, 64, DDIM, 1);
  wmma_gemm_kernel<<<8, 256, 0, stream>>>(G, HH, mlp_w2, HH, mlp_b2,
                                          Y, DDIM, 4, 16, HH, 0);
  ln2_head_kernel<<<NB, 256, 0, stream>>>(X, Y, ln2_w, ln2_b, head_w, head_b, pose);

  // 3x3 SVD orthogonalization + output assembly
  svd_head_kernel<<<1, 64, 0, stream>>>(pose, (float*)d_out);
}
